// Decoder_52235392254571
// MI455X (gfx1250) — compile-verified, matrix operand order checked by reading, not run on hardware
//
#include <hip/hip_runtime.h>
#include <hip/hip_bf16.h>
#include <math.h>

// ---------------------------------------------------------------------------
// Decoder (GRU + Bahdanau attention), CDNA5 / gfx1250, wave32, WMMA bf16.
// B=64 T=32 S=64 V=32000 E=512 H=1024 C=2048
// ---------------------------------------------------------------------------

typedef __bf16 bf16_t;
typedef __attribute__((ext_vector_type(16))) __bf16 bf16x16;
typedef __attribute__((ext_vector_type(8)))  __bf16 bf16x8;
typedef __attribute__((ext_vector_type(8)))  float  v8f;

#define Bb 64
#define Tt 32
#define Ss 64
#define Ee 512
#define Hh 1024
#define Cc 2048
#define G3 3072           // 3*H
#define PK 3584           // E + H + C (packed final-GEMM K)

// ---------------------------------------------------------------------------
// WMMA bf16 GEMM:  Y[M,N] = A[M,K] @ W[N,K]^T (+bias[N]) [+tanh]
// A row-major (lda), W row-major (ldw, K-contiguous), Y row-major f32 (ldy).
// Requires M%32==0, N%64==0, K%32==0 (true for every GEMM in this model).
// Each wave owns a 32x64 output tile: 2 A-frags x 4 B-frags -> 8 WMMAs/K-step,
// 21.8 FLOP per L2 byte (B-fragment traffic amortized over 32 rows).
// ---------------------------------------------------------------------------
__device__ __forceinline__ bf16x16 load_frag(const bf16_t* __restrict__ p, int half) {
    // CDNA5 16-bit A/B fragment: lane<16 holds K [0..7],[16..23];
    // lane>=16 holds K [8..15],[24..31]  (two contiguous 16B chunks).
    bf16x8 lo = *(const bf16x8*)(p + half * 8);
    bf16x8 hi = *(const bf16x8*)(p + 16 + half * 8);
    bf16x16 f;
#pragma unroll
    for (int i = 0; i < 8; ++i) { f[i] = lo[i]; f[i + 8] = hi[i]; }
    return f;
}

__global__ void k_gemm_bf16(const bf16_t* __restrict__ A, long lda,
                            const bf16_t* __restrict__ W, long ldw,
                            const float*  __restrict__ bias,
                            float* __restrict__ Y, long ldy,
                            int M, int N, int K, int epi /*0=none,1=tanh*/) {
    const int wavesPerBlk = blockDim.x >> 5;
    const int wave = blockIdx.x * wavesPerBlk + (threadIdx.x >> 5);
    const int tilesN = N >> 6;
    const int tiles  = (M >> 5) * tilesN;
    if (wave >= tiles) return;                 // wave-uniform: EXEC all-ones below
    const int tm = wave / tilesN;
    const int tn = wave % tilesN;
    const int lane = threadIdx.x & 31;
    const int half = lane >> 4;
    const int r    = lane & 15;

    const bf16_t* aRow0 = A + (long)(tm * 32 + r)      * lda;
    const bf16_t* aRow1 = A + (long)(tm * 32 + 16 + r) * lda;
    const bf16_t* wRow  = W + (long)(tn * 64 + r)      * ldw;

    v8f acc00 = {}, acc01 = {}, acc02 = {}, acc03 = {};
    v8f acc10 = {}, acc11 = {}, acc12 = {}, acc13 = {};
    for (int kb = 0; kb < K; kb += 32) {
        // max-locality prefetch -> WGP scope (pull next weight slab near)
        __builtin_prefetch(wRow + kb + 256, 0, 3);
        bf16x16 a0 = load_frag(aRow0 + kb, half);
        bf16x16 a1 = load_frag(aRow1 + kb, half);
        bf16x16 b0 = load_frag(wRow + kb, half);
        bf16x16 b1 = load_frag(wRow + 16 * ldw + kb, half);
        bf16x16 b2 = load_frag(wRow + 32 * ldw + kb, half);
        bf16x16 b3 = load_frag(wRow + 48 * ldw + kb, half);
        acc00 = __builtin_amdgcn_wmma_f32_16x16x32_bf16(false, a0, false, b0, (short)0, acc00, false, false);
        acc01 = __builtin_amdgcn_wmma_f32_16x16x32_bf16(false, a0, false, b1, (short)0, acc01, false, false);
        acc02 = __builtin_amdgcn_wmma_f32_16x16x32_bf16(false, a0, false, b2, (short)0, acc02, false, false);
        acc03 = __builtin_amdgcn_wmma_f32_16x16x32_bf16(false, a0, false, b3, (short)0, acc03, false, false);
        acc10 = __builtin_amdgcn_wmma_f32_16x16x32_bf16(false, a1, false, b0, (short)0, acc10, false, false);
        acc11 = __builtin_amdgcn_wmma_f32_16x16x32_bf16(false, a1, false, b1, (short)0, acc11, false, false);
        acc12 = __builtin_amdgcn_wmma_f32_16x16x32_bf16(false, a1, false, b2, (short)0, acc12, false, false);
        acc13 = __builtin_amdgcn_wmma_f32_16x16x32_bf16(false, a1, false, b3, (short)0, acc13, false, false);
    }

    // D layout: lane r holds column N=r; VGPR g holds row M = g + 8*half.
    float bv0 = bias ? bias[tn * 64 +  0 + r] : 0.f;
    float bv1 = bias ? bias[tn * 64 + 16 + r] : 0.f;
    float bv2 = bias ? bias[tn * 64 + 32 + r] : 0.f;
    float bv3 = bias ? bias[tn * 64 + 48 + r] : 0.f;
#pragma unroll
    for (int g = 0; g < 8; ++g) {
        long row0 = tm * 32 + g + 8 * half;
        long row1 = row0 + 16;
        float o00 = acc00[g] + bv0, o01 = acc01[g] + bv1, o02 = acc02[g] + bv2, o03 = acc03[g] + bv3;
        float o10 = acc10[g] + bv0, o11 = acc11[g] + bv1, o12 = acc12[g] + bv2, o13 = acc13[g] + bv3;
        if (epi == 1) {
            o00 = tanhf(o00); o01 = tanhf(o01); o02 = tanhf(o02); o03 = tanhf(o03);
            o10 = tanhf(o10); o11 = tanhf(o11); o12 = tanhf(o12); o13 = tanhf(o13);
        }
        float* y0 = Y + row0 * ldy + tn * 64 + r;
        float* y1 = Y + row1 * ldy + tn * 64 + r;
        y0[0] = o00; y0[16] = o01; y0[32] = o02; y0[48] = o03;
        y1[0] = o10; y1[16] = o11; y1[32] = o12; y1[48] = o13;
    }
}

// ---------------------------------------------------------------------------
// Prep / elementwise kernels
// ---------------------------------------------------------------------------
__global__ void k_cvt_bf16(const float* __restrict__ in, bf16_t* __restrict__ out, int n) {
    int i = blockIdx.x * blockDim.x + threadIdx.x;
    if (i < n) out[i] = (bf16_t)in[i];
}

// in[R,Cn] (row-major) -> out[Cn,R] bf16 (transpose)
__global__ void k_cvt_bf16_T(const float* __restrict__ in, bf16_t* __restrict__ out, int R, int Cn) {
    int i = blockIdx.x * blockDim.x + threadIdx.x;
    if (i < R * Cn) {
        int r = i % R, c = i / R;          // i = c*R + r (coalesced writes)
        out[i] = (bf16_t)in[(long)r * Cn + c];
    }
}

// packed final weights: Wfin[n, 0:512]=Wi[n], [512:1536]=Wh[n], [1536:3584]=Wc[n]
__global__ void k_pack_wfin(const float* __restrict__ Wi, const float* __restrict__ Wh,
                            const float* __restrict__ Wc, bf16_t* __restrict__ out) {
    int i = blockIdx.x * blockDim.x + threadIdx.x;
    if (i >= Ee * PK) return;
    int n = i / PK, k = i % PK;
    float v;
    if (k < Ee)            v = Wi[(long)n * Ee + k];
    else if (k < Ee + Hh)  v = Wh[(long)n * Hh + (k - Ee)];
    else                   v = Wc[(long)n * Cc + (k - Ee - Hh)];
    out[i] = (bf16_t)v;
}

__global__ void k_bias3(const float* a, const float* b, const float* c, float* o) {
    int i = blockIdx.x * blockDim.x + threadIdx.x;
    if (i < Ee) o[i] = a[i] + b[i] + c[i];
}

// embedding gather -> packed A columns [0:512) as bf16
__global__ void k_embed(const int* __restrict__ y, const float* __restrict__ emb,
                        bf16_t* __restrict__ packed) {
    int i = blockIdx.x * blockDim.x + threadIdx.x;    // over B*T*E
    if (i >= Bb * Tt * Ee) return;
    int bt = i / Ee, e = i % Ee;
    packed[(long)bt * PK + e] = (bf16_t)emb[(long)y[bt] * Ee + e];
}

__global__ void k_init_h(const float* __restrict__ hidden, float* hf, bf16_t* hb) {
    int i = blockIdx.x * blockDim.x + threadIdx.x;    // B*H
    if (i < Bb * Hh) { hf[i] = hidden[i]; hb[i] = (bf16_t)hidden[i]; }
}

// fused GRU gates: h = (1-z)*n + z*hprev ; biases already folded into gi/gh GEMMs
__global__ void k_gru(const float* __restrict__ gi, const float* __restrict__ gh,
                      const float* __restrict__ hprev,
                      float* __restrict__ hf, bf16_t* __restrict__ hb,
                      bf16_t* __restrict__ packed, int t) {
    int i = blockIdx.x * blockDim.x + threadIdx.x;    // B*H
    int b = i >> 10, j = i & (Hh - 1);
    const float* gib = gi + (long)b * G3;
    const float* ghb = gh + (long)b * G3;
    float r = 1.f / (1.f + expf(-(gib[j]          + ghb[j])));
    float z = 1.f / (1.f + expf(-(gib[Hh + j]     + ghb[Hh + j])));
    float n = tanhf(gib[2 * Hh + j] + r * ghb[2 * Hh + j]);
    float h = (1.f - z) * n + z * hprev[i];
    hf[i] = h;
    hb[i] = (bf16_t)h;
    if (packed) packed[((long)(b * Tt + t)) * PK + Ee + j] = (bf16_t)h;  // 'out' slice
}

// scores[b,s] = sum_h tanh(q[b,h] + cache[b,s,h]) * v[h] ; one wave32 per (b,s)
__global__ void k_scores(const float* __restrict__ q, const float* __restrict__ cache,
                         const float* __restrict__ v, float* __restrict__ scores) {
    int wid  = (blockIdx.x * blockDim.x + threadIdx.x) >> 5;
    int lane = threadIdx.x & 31;
    if (wid >= Bb * Ss) return;
    int b = wid >> 6;
    const float* qb = q + (long)b * Hh;
    const float* cb = cache + (long)wid * Hh;
    float acc = 0.f;
    for (int h = lane; h < Hh; h += 32) acc += tanhf(qb[h] + cb[h]) * v[h];
#pragma unroll
    for (int off = 16; off; off >>= 1) acc += __shfl_xor(acc, off, 32);
    if (lane == 0) scores[wid] = acc;
}

// masked softmax over S=64, one block of 64 threads per batch row (LDS tree)
__global__ void k_softmax(const float* __restrict__ scores, const int* __restrict__ mask,
                          float* __restrict__ probs) {
    __shared__ float sm[Ss];
    int b = blockIdx.x, s = threadIdx.x;
    float v = scores[b * Ss + s];
    if (mask[b * Ss + s] != 0) v = -3.0e38f;
    sm[s] = v; __syncthreads();
    for (int off = 32; off; off >>= 1) { if (s < off) sm[s] = fmaxf(sm[s], sm[s + off]); __syncthreads(); }
    float m = sm[0]; __syncthreads();
    float e = expf(v - m);
    sm[s] = e; __syncthreads();
    for (int off = 32; off; off >>= 1) { if (s < off) sm[s] += sm[s + off]; __syncthreads(); }
    probs[b * Ss + s] = e / sm[0];
}

// attn[b,c] = sum_s probs[b,s] * context[b,s,c]  -> bf16 (GEMM input + packed slice)
__global__ void k_attn(const float* __restrict__ probs, const float* __restrict__ ctx,
                       bf16_t* __restrict__ attn_b, bf16_t* __restrict__ packed, int t) {
    int i = blockIdx.x * blockDim.x + threadIdx.x;    // B*C
    int b = i >> 11, c = i & (Cc - 1);
    const float* pb = probs + b * Ss;
    const float* cb = ctx + ((long)b * Ss) * Cc + c;
    float acc = 0.f;
#pragma unroll 4
    for (int s = 0; s < Ss; ++s) acc += pb[s] * cb[(long)s * Cc];
    attn_b[i] = (bf16_t)acc;
    packed[((long)(b * Tt + t)) * PK + Ee + Hh + c] = (bf16_t)acc;
}

// ---------------------------------------------------------------------------
// Host side
// ---------------------------------------------------------------------------
static inline void launch_gemm(const bf16_t* A, long lda, const bf16_t* W, long ldw,
                               const float* bias, float* Y, long ldy,
                               int M, int N, int K, int epi, hipStream_t s) {
    int waves = (M >> 5) * (N >> 6);              // 32x64 wave tiles
    int blocks = (waves + 7) / 8;                 // 8 waves (256 threads) per block
    k_gemm_bf16<<<blocks, 256, 0, s>>>(A, lda, W, ldw, bias, Y, ldy, M, N, K, epi);
}

extern "C" void kernel_launch(void* const* d_in, const int* in_sizes, int n_in,
                              void* d_out, int out_size, void* d_ws, size_t ws_size,
                              hipStream_t stream) {
    (void)in_sizes; (void)n_in; (void)out_size; (void)ws_size;
    const int*   y      = (const int*)  d_in[0];
    const float* ctx    = (const float*)d_in[1];
    const int*   cmask  = (const int*)  d_in[2];
    const float* hidden = (const float*)d_in[3];
    const float* emb    = (const float*)d_in[4];
    const float* W_ih1  = (const float*)d_in[5];
    const float* W_hh1  = (const float*)d_in[6];
    const float* b_ih1  = (const float*)d_in[7];
    const float* b_hh1  = (const float*)d_in[8];
    const float* Wq     = (const float*)d_in[9];
    const float* bq     = (const float*)d_in[10];
    const float* Wk     = (const float*)d_in[11];
    const float* v_attn = (const float*)d_in[12];
    const float* W_ih2  = (const float*)d_in[13];
    const float* W_hh2  = (const float*)d_in[14];
    const float* b_ih2  = (const float*)d_in[15];
    const float* b_hh2  = (const float*)d_in[16];
    const float* Wi     = (const float*)d_in[17];
    const float* bi     = (const float*)d_in[18];
    const float* Wh     = (const float*)d_in[19];
    const float* bh     = (const float*)d_in[20];
    const float* Wc     = (const float*)d_in[21];
    const float* bc     = (const float*)d_in[22];
    float* out = (float*)d_out;                      // logits [B,T,E] then h_final [B,H]

    // workspace bump allocator (~90 MB total)
    char* w = (char*)d_ws;
    auto alloc = [&](size_t bytes) -> void* {
        void* p = (void*)w; w += (bytes + 255) & ~(size_t)255; return p;
    };
    bf16_t* packedA  = (bf16_t*)alloc((size_t)Bb * Tt * PK * 2);   // [x|out|attn]
    bf16_t* wfin     = (bf16_t*)alloc((size_t)Ee * PK * 2);
    float*  bfin     = (float*) alloc((size_t)Ee * 4);
    bf16_t* ctx_b    = (bf16_t*)alloc((size_t)Bb * Ss * Cc * 2);
    bf16_t* WkT_b    = (bf16_t*)alloc((size_t)Hh * Cc * 2);        // Wk^T [H,C]
    float*  cache    = (float*) alloc((size_t)Bb * Ss * Hh * 4);
    bf16_t* Wih1_b   = (bf16_t*)alloc((size_t)G3 * Ee * 2);
    bf16_t* Whh1_b   = (bf16_t*)alloc((size_t)G3 * Hh * 2);
    bf16_t* WqT_b    = (bf16_t*)alloc((size_t)Hh * Hh * 2);        // Wq^T [H,H]
    bf16_t* Wih2_b   = (bf16_t*)alloc((size_t)G3 * Cc * 2);
    bf16_t* Whh2_b   = (bf16_t*)alloc((size_t)G3 * Hh * 2);
    float*  h_f      = (float*) alloc((size_t)Bb * Hh * 4);
    bf16_t* h_b      = (bf16_t*)alloc((size_t)Bb * Hh * 2);
    float*  h1_f     = (float*) alloc((size_t)Bb * Hh * 4);
    bf16_t* h1_b     = (bf16_t*)alloc((size_t)Bb * Hh * 2);
    float*  qbuf     = (float*) alloc((size_t)Bb * Hh * 4);
    float*  scores   = (float*) alloc((size_t)Bb * Ss * 4);
    float*  probs    = (float*) alloc((size_t)Bb * Ss * 4);
    bf16_t* attn_b   = (bf16_t*)alloc((size_t)Bb * Cc * 2);
    float*  gi       = (float*) alloc((size_t)Bb * G3 * 4);
    float*  gh       = (float*) alloc((size_t)Bb * G3 * 4);

    const int TPB = 256;
    auto grid = [](long n, int tpb) { return (int)((n + tpb - 1) / tpb); };

    // ---- one-time prep ----
    k_cvt_bf16<<<grid((long)Bb*Ss*Cc, TPB), TPB, 0, stream>>>(ctx, ctx_b, Bb*Ss*Cc);
    k_cvt_bf16_T<<<grid((long)Cc*Hh, TPB), TPB, 0, stream>>>(Wk, WkT_b, Cc, Hh);
    k_cvt_bf16<<<grid((long)G3*Ee, TPB), TPB, 0, stream>>>(W_ih1, Wih1_b, G3*Ee);
    k_cvt_bf16<<<grid((long)G3*Hh, TPB), TPB, 0, stream>>>(W_hh1, Whh1_b, G3*Hh);
    k_cvt_bf16_T<<<grid((long)Hh*Hh, TPB), TPB, 0, stream>>>(Wq, WqT_b, Hh, Hh);
    k_cvt_bf16<<<grid((long)G3*Cc, TPB), TPB, 0, stream>>>(W_ih2, Wih2_b, G3*Cc);
    k_cvt_bf16<<<grid((long)G3*Hh, TPB), TPB, 0, stream>>>(W_hh2, Whh2_b, G3*Hh);
    k_pack_wfin<<<grid((long)Ee*PK, TPB), TPB, 0, stream>>>(Wi, Wh, Wc, wfin);
    k_bias3<<<grid(Ee, TPB), TPB, 0, stream>>>(bi, bh, bc, bfin);
    k_embed<<<grid((long)Bb*Tt*Ee, TPB), TPB, 0, stream>>>(y, emb, packedA);
    k_init_h<<<grid((long)Bb*Hh, TPB), TPB, 0, stream>>>(hidden, h_f, h_b);

    // ---- key cache: cache[B*S,H] = context[B*S,C] @ Wk  (WMMA) ----
    launch_gemm(ctx_b, Cc, WkT_b, Cc, nullptr, cache, Hh, Bb*Ss, Hh, Cc, 0, stream);

    // ---- recurrent scan over T ----
    for (int t = 0; t < Tt; ++t) {
        const bf16_t* xt = packedA + (long)t * PK;      // row b at (b*T+t)*PK, cols [0,E)
        // GRU1 gates
        launch_gemm(xt,  (long)Tt * PK, Wih1_b, Ee, b_ih1, gi, G3, Bb, G3, Ee, 0, stream);
        launch_gemm(h_b, Hh,            Whh1_b, Hh, b_hh1, gh, G3, Bb, G3, Hh, 0, stream);
        k_gru<<<grid((long)Bb*Hh, TPB), TPB, 0, stream>>>(gi, gh, h_f, h1_f, h1_b, nullptr, t);
        // q = h1 @ Wq + bq
        launch_gemm(h1_b, Hh, WqT_b, Hh, bq, qbuf, Hh, Bb, Hh, Hh, 0, stream);
        // attention
        k_scores<<<grid((long)Bb*Ss*32, TPB), TPB, 0, stream>>>(qbuf, cache, v_attn, scores);
        k_softmax<<<Bb, Ss, 0, stream>>>(scores, cmask, probs);
        k_attn<<<grid((long)Bb*Cc, TPB), TPB, 0, stream>>>(probs, ctx, attn_b, packedA, t);
        // GRU2 gates -> new state (also written into packed 'out' slice)
        launch_gemm(attn_b, Cc, Wih2_b, Cc, b_ih2, gi, G3, Bb, G3, Cc, 0, stream);
        launch_gemm(h1_b,  Hh, Whh2_b, Hh, b_hh2, gh, G3, Bb, G3, Hh, 0, stream);
        k_gru<<<grid((long)Bb*Hh, TPB), TPB, 0, stream>>>(gi, gh, h1_f, h_f, h_b, packedA, t);
    }

    // ---- logits = tanh([x|out|attn] @ [Wi|Wh|Wc]^T + (bi+bh+bc)) ----
    launch_gemm(packedA, PK, wfin, PK, bfin, out, Ee, Bb*Tt, Ee, PK, 1, stream);

    // ---- h_final ----
    hipMemcpyAsync(out + (size_t)Bb * Tt * Ee, h_f, (size_t)Bb * Hh * sizeof(float),
                   hipMemcpyDeviceToDevice, stream);
}
